// SynthText_CRNN_89893665506107
// MI455X (gfx1250) — matrix operand order repaired and checked by reading, not compiled, not run on hardware
//
#include <hip/hip_runtime.h>
#include <hip/hip_bf16.h>
#include <math.h>

// ---------------------------------------------------------------------------
// CDNA5 (gfx1250) wave32 WMMA helpers: D(16x16 f32) = A(16x32 f16) x B(32x16 f16) + C
// ---------------------------------------------------------------------------
typedef __attribute__((ext_vector_type(16))) _Float16 v16h;
typedef __attribute__((ext_vector_type(8)))  _Float16 v8h;
typedef __attribute__((ext_vector_type(8)))  float    v8f;

union AFrag { v16h v; _Float16 h[16]; };
union CFrag { v8f  v; float    f[8];  };

// A-matrix (16x32 f16) layout: lane<16 rows M=lane, halves j -> K = kc + 16*hi + j:
// one contiguous 16-half (32B) run per lane (matches B).
__device__ __forceinline__ AFrag load_a_f16(const _Float16* __restrict__ p, int hi) {
  AFrag a;
  *(v8h*)&a.h[0] = *(const v8h*)(p + 16 * hi);
  *(v8h*)&a.h[8] = *(const v8h*)(p + 16 * hi + 8);
  return a;
}
// B-matrix (32x16 f16): N = lane&15, halves j -> K = kc + 16*hi + j:
// one contiguous 16-half (32B) run.
__device__ __forceinline__ AFrag load_b_f16(const _Float16* __restrict__ p, int hi) {
  AFrag b;
  const _Float16* q = p + 16 * hi;
  *(v8h*)&b.h[0] = *(const v8h*)(q);
  *(v8h*)&b.h[8] = *(const v8h*)(q + 8);
  return b;
}

__device__ __forceinline__ v8f wmma32(const AFrag& a, const AFrag& b, v8f acc) {
  return __builtin_amdgcn_wmma_f32_16x16x32_f16(false, a.v, false, b.v,
                                                (short)0, acc, false, false);
}

__device__ __forceinline__ float sigm(float x) { return 1.0f / (1.0f + expf(-x)); }

// ---------------------------------------------------------------------------
// Conversions (f32 -> f16), with optional padding (zero fill)
// ---------------------------------------------------------------------------
__global__ void cvt_f16(const float* __restrict__ in, _Float16* __restrict__ out,
                        long long n) {
  for (long long i = (long long)blockIdx.x * 256 + threadIdx.x; i < n;
       i += (long long)gridDim.x * 256)
    out[i] = (_Float16)in[i];
}

// conv weights [Co][K] -> [Co][Kp], zero-padded k in [K,Kp)
__global__ void cvt_convw(const float* __restrict__ in, _Float16* __restrict__ out,
                          int Co, int K, int Kp) {
  int n = Co * Kp;
  for (int idx = blockIdx.x * 256 + threadIdx.x; idx < n; idx += gridDim.x * 256) {
    int k = idx % Kp;
    int oc = idx / Kp;
    out[idx] = (k < K) ? (_Float16)in[(long long)oc * K + k] : (_Float16)0.0f;
  }
}

// row-padded matrix [rows][cols] -> [rowsPad][cols], zero rows beyond `rows`
__global__ void cvt_rowpad(const float* __restrict__ in, _Float16* __restrict__ out,
                           int rows, int cols, int rowsPad) {
  int n = rowsPad * cols;
  for (int idx = blockIdx.x * 256 + threadIdx.x; idx < n; idx += gridDim.x * 256) {
    int r = idx / cols;
    out[idx] = (r < rows) ? (_Float16)in[idx] : (_Float16)0.0f;
  }
}

// ---------------------------------------------------------------------------
// Implicit-GEMM conv (stride 1) + bias + ReLU.  Input activations are f16.
// Block = 256 threads = 8 waves; each wave computes 64 oc x 16 positions:
// the scalar im2col B-fragment is built once per k-chunk and reused by
// 4 WMMAs (A = weights, vector b128 loads).  KHW/KW are compile-time so the
// k-decomposition divisions become mul-shift.
//   in16: f16 [16][Ci][H][W]   w16: f16 [Co][Kp] (zero-padded)   out: f32
// ---------------------------------------------------------------------------
template <int KHW, int KW>
__global__ void conv_wmma(const _Float16* __restrict__ in16,
                          const _Float16* __restrict__ w16,
                          const float* __restrict__ bias, float* __restrict__ out,
                          int Ci, int H, int Wd, int Co, int pad,
                          int Ho, int Wo, int Kp) {
  const int b    = blockIdx.z;
  const int oc0  = blockIdx.x * 64;
  const int wave = threadIdx.x >> 5;
  const int p0   = (blockIdx.y * 8 + wave) * 16;
  const int P    = Ho * Wo;
  if (p0 >= P) return;
  const int K    = Ci * KHW;
  const int lane = threadIdx.x & 31;
  const int hi   = (lane >> 4) & 1;
  const int ml   = lane & 15;

  CFrag acc[4];
#pragma unroll
  for (int i = 0; i < 4; ++i)
#pragma unroll
    for (int r = 0; r < 8; ++r) acc[i].f[r] = 0.0f;

  const int pos = p0 + ml;
  const int ho = (pos < P) ? pos / Wo : 0;
  const int wo = (pos < P) ? pos - ho * Wo : 0;
  const bool posok = pos < P;

  for (int kc = 0; kc < Kp; kc += 32) {
    AFrag bm;  // im2col: N = position, K gathered (masked)
#pragma unroll
    for (int j = 0; j < 16; ++j) {
      int k = kc + j + hi * 16;
      _Float16 val = (_Float16)0.0f;
      if (posok && k < K) {
        int ci = k / KHW;           // compile-time divisor
        int rem = k - ci * KHW;
        int r = rem / KW;           // compile-time divisor
        int s = rem - r * KW;
        int hin = ho - pad + r, win = wo - pad + s;
        if (hin >= 0 && hin < H && win >= 0 && win < Wd)
          val = in16[(((long long)b * Ci + ci) * H + hin) * Wd + win];
      }
      bm.h[j] = val;
    }
#pragma unroll
    for (int i = 0; i < 4; ++i) {
      AFrag a = load_a_f16(w16 + (long long)(oc0 + 16 * i + ml) * Kp + kc, hi);
      acc[i].v = wmma32(a, bm, acc[i].v);
    }
  }
#pragma unroll
  for (int i = 0; i < 4; ++i) {
#pragma unroll
    for (int r = 0; r < 8; ++r) {
      int oc = oc0 + 16 * i + r + hi * 8;
      if (posok) {
        float v = fmaxf(acc[i].f[r] + bias[oc], 0.0f);
        out[(((long long)b * Co + oc) * Ho + ho) * Wo + wo] = v;
      }
    }
  }
}

// ---------------------------------------------------------------------------
// 2D max pool: f32 in (conv output) -> f16 out (next conv's input)
// ---------------------------------------------------------------------------
__global__ void maxpool2d(const float* __restrict__ in, _Float16* __restrict__ out,
                          int BC, int H, int W, int kh, int kw, int sh, int sw,
                          int Hp, int Wp) {
  long long n = (long long)BC * Hp * Wp;
  for (long long idx = (long long)blockIdx.x * 256 + threadIdx.x; idx < n;
       idx += (long long)gridDim.x * 256) {
    int wp = (int)(idx % Wp);
    long long r = idx / Wp;
    int hp = (int)(r % Hp);
    long long bc = r / Hp;
    float m = -3.4e38f;
    for (int i = 0; i < kh; ++i)
      for (int j = 0; j < kw; ++j)
        m = fmaxf(m, in[(bc * H + hp * sh + i) * W + wp * sw + j]);
    out[idx] = (_Float16)m;
  }
}

// ---------------------------------------------------------------------------
// [16][512][1][61] f32 -> [16][61][512] f16 (feeds first LSTM-layer GEMM)
// ---------------------------------------------------------------------------
__global__ void squeeze_permute(const float* __restrict__ in, _Float16* __restrict__ out) {
  const int n = 16 * 61 * 512;
  for (int idx = blockIdx.x * 256 + threadIdx.x; idx < n; idx += gridDim.x * 256) {
    int c = idx & 511;
    int r = idx >> 9;
    int t = r % 61;
    int b = r / 61;
    out[idx] = (_Float16)in[(b * 512 + c) * 61 + t];
  }
}

// ---------------------------------------------------------------------------
// GEMM: C[M][N] = A_f16[M][lda] * W_f16[N][ldw]^T (+ b0[n] + b1[n]).
// 256-thread blocks = 8 waves; wave owns a 16M x 64N strip (A-fragment
// reused by 4 WMMAs).  Requires N%64==0, K%32==0.
// ---------------------------------------------------------------------------
__global__ void gemm_aw(const _Float16* __restrict__ A, int lda,
                        const _Float16* __restrict__ W, int ldw,
                        const float* __restrict__ b0, const float* __restrict__ b1,
                        float* __restrict__ C, int ldc, int M, int K) {
  const int n0 = blockIdx.x * 64;
  const int wave = threadIdx.x >> 5;
  const int m0 = (blockIdx.y * 8 + wave) * 16;
  if (m0 >= M) return;
  const int lane = threadIdx.x & 31;
  const int hi = (lane >> 4) & 1;
  const int ml = lane & 15;

  CFrag acc[4];
#pragma unroll
  for (int i = 0; i < 4; ++i)
#pragma unroll
    for (int r = 0; r < 8; ++r) acc[i].f[r] = 0.0f;

  for (int kc = 0; kc < K; kc += 32) {
    AFrag a = load_a_f16(A + (long long)(m0 + ml) * lda + kc, hi);
#pragma unroll
    for (int i = 0; i < 4; ++i) {
      AFrag bm = load_b_f16(W + (long long)(n0 + 16 * i + ml) * ldw + kc, hi);
      acc[i].v = wmma32(a, bm, acc[i].v);
    }
  }
#pragma unroll
  for (int i = 0; i < 4; ++i) {
#pragma unroll
    for (int r = 0; r < 8; ++r) {
      int row = m0 + r + hi * 8;
      int col = n0 + 16 * i + ml;
      float v = acc[i].f[r];
      if (b0) v += b0[col];
      if (b1) v += b1[col];
      C[(long long)row * ldc + col] = v;
    }
  }
}

// ---------------------------------------------------------------------------
// Persistent LSTM direction kernel. grid.x = 2 (0 = fwd, 1 = bwd).
// 512 threads = 16 waves; wave w owns hidden chunk [16w,16w+16) of H=256 and
// computes all 4 gate tiles (M=16 batch).  xg holds x*Wih^T+bih+bhh,
// laid out [b*T + t][1024].  Writes f32 seq (+ f16 copy for next GEMM).
// Cell state in registers; h state in LDS as f16 (WMMA A operand).
// ---------------------------------------------------------------------------
__global__ void lstm_dir(const float* __restrict__ xgF, const float* __restrict__ xgB,
                         const _Float16* __restrict__ whhF,
                         const _Float16* __restrict__ whhB,
                         float* __restrict__ outSeq, _Float16* __restrict__ outSeq16,
                         int T) {
  const int dir = blockIdx.x;
  const float* __restrict__ xg = dir ? xgB : xgF;
  const _Float16* __restrict__ whh = dir ? whhB : whhF;
  const int dirOff = dir * 256;
  const int tid = threadIdx.x;
  const int wave = tid >> 5;
  const int lane = tid & 31;
  const int hi = (lane >> 4) & 1;
  const int ml = lane & 15;

  __shared__ __align__(16) _Float16 sh_hf[16 * 256];  // h state (f16) [batch][hidden]
  for (int i = tid; i < 16 * 256; i += 512) sh_hf[i] = (_Float16)0.0f;
  CFrag c;
#pragma unroll
  for (int r = 0; r < 8; ++r) c.f[r] = 0.0f;
  __syncthreads();

  for (int tt = 0; tt < T; ++tt) {
    const int t = dir ? (T - 1 - tt) : tt;

    // prefetch next step's xg slice while this step's WMMAs run
    if (tt + 1 < T) {
      int tn = dir ? (T - 2 - tt) : (tt + 1);
      long long pb = ((long long)(hi * 8) * T + tn) * 1024;
      __builtin_prefetch(&xg[pb + wave * 16 + ml], 0, 1);
    }

    CFrag acc[4];
#pragma unroll
    for (int g = 0; g < 4; ++g)
#pragma unroll
      for (int r = 0; r < 8; ++r) acc[g].f[r] = 0.0f;

    for (int kc = 0; kc < 256; kc += 32) {
      AFrag a = load_a_f16(sh_hf + ml * 256 + kc, hi);  // A = h
#pragma unroll
      for (int g = 0; g < 4; ++g) {
        int row = g * 256 + wave * 16 + ml;
        AFrag bm = load_b_f16(whh + (long long)row * 256 + kc, hi);
        acc[g].v = wmma32(a, bm, acc[g].v);
      }
    }
    __syncthreads();  // all reads of sh_hf done

#pragma unroll
    for (int r = 0; r < 8; ++r) {
      int m = r + hi * 8;          // batch
      int hidx = wave * 16 + ml;   // hidden index
      long long base = ((long long)m * T + t) * 1024;
      float gi = acc[0].f[r] + xg[base + hidx];
      float gf = acc[1].f[r] + xg[base + 256 + hidx];
      float gg = acc[2].f[r] + xg[base + 512 + hidx];
      float go = acc[3].f[r] + xg[base + 768 + hidx];
      float cN = sigm(gf) * c.f[r] + sigm(gi) * tanhf(gg);
      float h = sigm(go) * tanhf(cN);
      c.f[r] = cN;
      sh_hf[m * 256 + hidx] = (_Float16)h;
      long long o = ((long long)m * T + t) * 512 + dirOff + hidx;
      outSeq[o] = h;
      outSeq16[o] = (_Float16)h;
    }
    __syncthreads();
  }
}

// ---------------------------------------------------------------------------
// Persistent attention decoder, one block of 512 threads (16 waves), 25 steps.
//   enc [976][256] f32, seq [976][512] f32, dense padded to [112][512].
// ---------------------------------------------------------------------------
__global__ void attn_decoder(const float* __restrict__ enc, const float* __restrict__ seq,
                             const _Float16* __restrict__ decp,  // [256][512]
                             const float* __restrict__ decp_b, const float* __restrict__ vw,
                             const _Float16* __restrict__ wih,   // [2048][512]
                             const _Float16* __restrict__ whh,   // [2048][512]
                             const float* __restrict__ bih, const float* __restrict__ bhh,
                             const _Float16* __restrict__ wd,    // [112][512] (padded)
                             const float* __restrict__ bd, float* __restrict__ out,
                             int T) {
  const int tid = threadIdx.x;
  const int wave = tid >> 5;
  const int lane = tid & 31;
  const int hi = (lane >> 4) & 1;
  const int ml = lane & 15;

  __shared__ __align__(16) _Float16 sh_hf[16 * 512];   // h (f16)
  __shared__ __align__(16) _Float16 sh_ctx[16 * 512];  // ctx (f16)
  __shared__ float sh_dec[16 * 256];
  __shared__ float sh_aw[16 * 64];

  for (int i = tid; i < 16 * 512; i += 512) sh_hf[i] = (_Float16)0.0f;
  CFrag c[2];
#pragma unroll
  for (int cc = 0; cc < 2; ++cc)
#pragma unroll
    for (int r = 0; r < 8; ++r) c[cc].f[r] = 0.0f;
  __syncthreads();

  for (int step = 0; step < 25; ++step) {
    // ---- A: dec = h * dec_proj^T + b  (wave w -> N tile w of 256) ----
    {
      CFrag acc;
#pragma unroll
      for (int r = 0; r < 8; ++r) acc.f[r] = 0.0f;
      int n0 = wave * 16;
      for (int kc = 0; kc < 512; kc += 32) {
        AFrag a = load_a_f16(sh_hf + ml * 512 + kc, hi);
        AFrag bm = load_b_f16(decp + (long long)(n0 + ml) * 512 + kc, hi);
        acc.v = wmma32(a, bm, acc.v);
      }
#pragma unroll
      for (int r = 0; r < 8; ++r) {
        int m = r + hi * 8;
        sh_dec[m * 256 + n0 + ml] = acc.f[r] + decp_b[n0 + ml];
      }
    }
    __syncthreads();

    // ---- B: scores[b][t] = sum_a tanh(enc + dec) * v ----
    for (int idx = tid; idx < 16 * T; idx += 512) {
      int b = idx / T, t = idx - b * T;
      const float* e = enc + ((long long)b * T + t) * 256;
      float s = 0.0f;
      for (int a0 = 0; a0 < 256; ++a0) s += tanhf(e[a0] + sh_dec[b * 256 + a0]) * vw[a0];
      sh_aw[b * 64 + t] = s;
    }
    __syncthreads();

    // ---- C: softmax over t (wave w handles batch w) ----
    {
      int b = wave;
      float mx = -3.4e38f;
      for (int t = lane; t < T; t += 32) mx = fmaxf(mx, sh_aw[b * 64 + t]);
      for (int o = 16; o > 0; o >>= 1) mx = fmaxf(mx, __shfl_xor(mx, o, 32));
      float sm = 0.0f;
      for (int t = lane; t < T; t += 32) {
        float ev = expf(sh_aw[b * 64 + t] - mx);
        sh_aw[b * 64 + t] = ev;
        sm += ev;
      }
      for (int o = 16; o > 0; o >>= 1) sm += __shfl_xor(sm, o, 32);
      float inv = 1.0f / sm;
      for (int t = lane; t < T; t += 32) sh_aw[b * 64 + t] *= inv;
    }
    __syncthreads();

    // ---- D: ctx[b][e] = sum_t aw * seq ----
    for (int idx = tid; idx < 16 * 512; idx += 512) {
      int b = idx >> 9, e = idx & 511;
      float s = 0.0f;
      for (int t = 0; t < T; ++t)
        s += sh_aw[b * 64 + t] * seq[((long long)b * T + t) * 512 + e];
      sh_ctx[b * 512 + e] = (_Float16)s;
    }
    __syncthreads();

    // ---- E: g = ctx*Wih^T + h*Whh^T + biases; LSTM cell update ----
    CFrag acc[2][4];
#pragma unroll
    for (int cc = 0; cc < 2; ++cc)
#pragma unroll
      for (int g = 0; g < 4; ++g)
#pragma unroll
        for (int r = 0; r < 8; ++r) acc[cc][g].f[r] = 0.0f;

    for (int cc = 0; cc < 2; ++cc) {
      int chunk = wave * 2 + cc;  // hidden chunk of 16 within 512
      for (int kc = 0; kc < 512; kc += 32) {
        AFrag ax = load_a_f16(sh_ctx + ml * 512 + kc, hi);
        AFrag ah = load_a_f16(sh_hf + ml * 512 + kc, hi);
#pragma unroll
        for (int g = 0; g < 4; ++g) {
          int row = g * 512 + chunk * 16 + ml;
          AFrag b1 = load_b_f16(wih + (long long)row * 512 + kc, hi);
          AFrag b2 = load_b_f16(whh + (long long)row * 512 + kc, hi);
          acc[cc][g].v = wmma32(ax, b1, acc[cc][g].v);
          acc[cc][g].v = wmma32(ah, b2, acc[cc][g].v);
        }
      }
    }
    __syncthreads();  // reads of sh_hf/sh_ctx done before writing sh_hf

    for (int cc = 0; cc < 2; ++cc) {
      int chunk = wave * 2 + cc;
      int hidx = chunk * 16 + ml;
#pragma unroll
      for (int r = 0; r < 8; ++r) {
        int m = r + hi * 8;
        float gi = acc[cc][0].f[r] + bih[hidx] + bhh[hidx];
        float gf = acc[cc][1].f[r] + bih[512 + hidx] + bhh[512 + hidx];
        float gg = acc[cc][2].f[r] + bih[1024 + hidx] + bhh[1024 + hidx];
        float go = acc[cc][3].f[r] + bih[1536 + hidx] + bhh[1536 + hidx];
        float cN = sigm(gf) * c[cc].f[r] + sigm(gi) * tanhf(gg);
        float h = sigm(go) * tanhf(cN);
        c[cc].f[r] = cN;
        sh_hf[m * 512 + hidx] = (_Float16)h;
      }
    }
    __syncthreads();

    // ---- F: logits = h * dense^T + b  (7 N tiles cover 99 cols; wd padded) ----
    if (wave < 7) {
      CFrag acc2;
#pragma unroll
      for (int r = 0; r < 8; ++r) acc2.f[r] = 0.0f;
      int n0 = wave * 16;
      for (int kc = 0; kc < 512; kc += 32) {
        AFrag a = load_a_f16(sh_hf + ml * 512 + kc, hi);
        AFrag bm = load_b_f16(wd + (long long)(n0 + ml) * 512 + kc, hi);
        acc2.v = wmma32(a, bm, acc2.v);
      }
#pragma unroll
      for (int r = 0; r < 8; ++r) {
        int m = r + hi * 8;
        int col = n0 + ml;
        if (col < 99) out[((long long)step * 16 + m) * 99 + col] = acc2.f[r] + bd[col];
      }
    }
    __syncthreads();
  }
}

// ---------------------------------------------------------------------------
// Host-side pipeline
// ---------------------------------------------------------------------------
static const int Ci_[7]  = {1, 64, 128, 256, 512, 512, 512};
static const int Co_[7]  = {64, 128, 256, 512, 512, 512, 512};
static const int Kk_[7]  = {3, 3, 3, 3, 3, 3, 2};
static const int pad_[7] = {1, 1, 1, 1, 1, 1, 0};
static const int Hin_[7] = {128, 64, 32, 16, 8, 4, 2};
static const int Win_[7] = {512, 256, 128, 127, 126, 63, 62};
static const int Ho_[7]  = {128, 64, 32, 16, 8, 4, 1};
static const int Wo_[7]  = {512, 256, 128, 127, 126, 63, 61};
static const int ssw_[6] = {2, 2, 1, 1, 2, 1};  // pool stride-W (stride-H 2, kernel 2x2)
static const int Hp_[6]  = {64, 32, 16, 8, 4, 2};
static const int Wp_[6]  = {256, 128, 127, 126, 63, 62};

extern "C" void kernel_launch(void* const* d_in, const int* in_sizes, int n_in,
                              void* d_out, int out_size, void* d_ws, size_t ws_size,
                              hipStream_t stream) {
  (void)in_sizes; (void)n_in; (void)out_size; (void)ws_size;
  const float* x = (const float*)d_in[0];
  const float* convw[7];
  const float* convb[7];
  for (int i = 0; i < 7; ++i) {
    convw[i] = (const float*)d_in[1 + 2 * i];
    convb[i] = (const float*)d_in[2 + 2 * i];
  }
  const float* lstm_wih = (const float*)d_in[15];
  const float* lstm_whh = (const float*)d_in[16];
  const float* lstm_bih = (const float*)d_in[17];
  const float* lstm_bhh = (const float*)d_in[18];
  const float* encp_w = (const float*)d_in[19];
  const float* encp_b = (const float*)d_in[20];
  const float* decp_w = (const float*)d_in[21];
  const float* decp_b = (const float*)d_in[22];
  const float* v_w    = (const float*)d_in[23];
  const float* dwih   = (const float*)d_in[24];
  const float* dwhh   = (const float*)d_in[25];
  const float* dbih   = (const float*)d_in[26];
  const float* dbhh   = (const float*)d_in[27];
  const float* dense_w = (const float*)d_in[28];
  const float* dense_b = (const float*)d_in[29];
  float* out = (float*)d_out;

  // workspace carve-up
  char* base = (char*)d_ws;
  size_t off = 0;
  auto take = [&](size_t bytes) -> char* {
    char* p = base + off;
    off += (bytes + 255) & ~(size_t)255;
    return p;
  };
  float*    convbuf = (float*)take((size_t)67108864 * 4);    // max pre-pool act (L0), f32
  _Float16* poolbuf = (_Float16*)take((size_t)16777216 * 2); // max pooled act (L0), f16
  _Float16* x16     = (_Float16*)take((size_t)1048576 * 2);  // input as f16
  _Float16* seq016  = (_Float16*)take((size_t)499712 * 2);   // 16*61*512 f16
  float*    seqA    = (float*)take((size_t)499712 * 4);
  _Float16* seqA16  = (_Float16*)take((size_t)499712 * 2);
  float*    seqB    = (float*)take((size_t)499712 * 4);
  _Float16* seqB16  = (_Float16*)take((size_t)499712 * 2);
  float*    xg0     = (float*)take((size_t)999424 * 4);      // 976*1024
  float*    xg1     = (float*)take((size_t)999424 * 4);
  float*    enc     = (float*)take((size_t)249856 * 4);      // 976*256
  int Kp_[7];
  _Float16* cw16[7];
  for (int i = 0; i < 7; ++i) {
    int K = Ci_[i] * Kk_[i] * Kk_[i];
    Kp_[i] = (K + 31) & ~31;  // pad K to multiple of 32 (only L0: 9 -> 32)
    cw16[i] = (_Float16*)take((size_t)Co_[i] * Kp_[i] * 2);
  }
  _Float16* wih16  = (_Float16*)take((size_t)8 * 1024 * 512 * 2);
  _Float16* whh16  = (_Float16*)take((size_t)8 * 1024 * 256 * 2);
  _Float16* encp16 = (_Float16*)take((size_t)256 * 512 * 2);
  _Float16* decp16 = (_Float16*)take((size_t)256 * 512 * 2);
  _Float16* dwih16 = (_Float16*)take((size_t)2048 * 512 * 2);
  _Float16* dwhh16 = (_Float16*)take((size_t)2048 * 512 * 2);
  _Float16* dense16 = (_Float16*)take((size_t)112 * 512 * 2);  // row-padded 99->112

  auto cvt = [&](const float* src, _Float16* dst, long long n) {
    int blocks = (int)((n + 255) / 256);
    if (blocks > 2048) blocks = 2048;
    cvt_f16<<<blocks, 256, 0, stream>>>(src, dst, n);
  };
  for (int i = 0; i < 7; ++i) {
    int K = Ci_[i] * Kk_[i] * Kk_[i];
    int n = Co_[i] * Kp_[i];
    int blocks = (n + 255) / 256;
    if (blocks > 2048) blocks = 2048;
    cvt_convw<<<blocks, 256, 0, stream>>>(convw[i], cw16[i], Co_[i], K, Kp_[i]);
  }
  cvt(lstm_wih, wih16, 8LL * 1024 * 512);
  cvt(lstm_whh, whh16, 8LL * 1024 * 256);
  cvt(encp_w, encp16, 256LL * 512);
  cvt(decp_w, decp16, 256LL * 512);
  cvt(dwih, dwih16, 2048LL * 512);
  cvt(dwhh, dwhh16, 2048LL * 512);
  cvt_rowpad<<<224, 256, 0, stream>>>(dense_w, dense16, 99, 512, 112);
  cvt(x, x16, 1048576LL);  // input image -> f16

  // conv / pool chain (conv reads f16, writes f32; pool reads f32, writes f16)
  const _Float16* cur = x16;
  for (int l = 0; l < 7; ++l) {
    int P = Ho_[l] * Wo_[l];
    dim3 grid((Co_[l] + 63) / 64, (P + 127) / 128, 16);
    if (Kk_[l] == 3)
      conv_wmma<9, 3><<<grid, 256, 0, stream>>>(cur, cw16[l], convb[l], convbuf,
                                                Ci_[l], Hin_[l], Win_[l], Co_[l],
                                                pad_[l], Ho_[l], Wo_[l], Kp_[l]);
    else
      conv_wmma<4, 2><<<grid, 256, 0, stream>>>(cur, cw16[l], convb[l], convbuf,
                                                Ci_[l], Hin_[l], Win_[l], Co_[l],
                                                pad_[l], Ho_[l], Wo_[l], Kp_[l]);
    if (l < 6) {
      long long n = 16LL * Co_[l] * Hp_[l] * Wp_[l];
      int blocks = (int)((n + 255) / 256);
      if (blocks > 4096) blocks = 4096;
      maxpool2d<<<blocks, 256, 0, stream>>>(convbuf, poolbuf, 16 * Co_[l],
                                            Ho_[l], Wo_[l], 2, 2, 2, ssw_[l],
                                            Hp_[l], Wp_[l]);
      cur = poolbuf;
    }
  }
  squeeze_permute<<<512, 256, 0, stream>>>(convbuf, seq016);

  // 4 bidirectional LSTM layers (2 stacks of 2)
  const _Float16* lin16[4] = {seq016, seqA16, seqB16, seqA16};
  float* lout[4] = {seqA, seqB, seqA, seqB};
  _Float16* lout16[4] = {seqA16, seqB16, seqA16, seqB16};
  const int slots[4] = {0, 2, 4, 6};
  for (int L = 0; L < 4; ++L) {
    for (int d = 0; d < 2; ++d) {
      int s = slots[L] + d;
      gemm_aw<<<dim3(16, 8), 256, 0, stream>>>(
          lin16[L], 512, wih16 + (size_t)s * 524288, 512,
          lstm_bih + (size_t)s * 1024, lstm_bhh + (size_t)s * 1024,
          d ? xg1 : xg0, 1024, 976, 512);
    }
    lstm_dir<<<2, 512, 0, stream>>>(xg0, xg1,
                                    whh16 + (size_t)slots[L] * 262144,
                                    whh16 + (size_t)(slots[L] + 1) * 262144,
                                    lout[L], lout16[L], 61);
  }

  // attention: enc projection (h-independent), then persistent decoder
  gemm_aw<<<dim3(4, 8), 256, 0, stream>>>(seqB16, 512, encp16, 512, encp_b,
                                          (const float*)nullptr, enc, 256, 976, 512);
  attn_decoder<<<1, 512, 0, stream>>>(enc, seqB, decp16, decp_b, v_w, dwih16,
                                      dwhh16, dbih, dbhh, dense16, dense_b,
                                      out, 61);
}
